// Net_2284922601976
// MI455X (gfx1250) — matrix operand order repaired and checked by reading, not compiled, not run on hardware
//
#include <hip/hip_runtime.h>
#include <hip/hip_bf16.h>

#define Nn 100000
#define Ee 800000
#define INc 64
#define Hc 128
#define Kc 9
#define EPSf 1e-15f

// d_out offsets (floats)
#define OFF_OUT 0
#define OFF_MC  1152
#define OFF_O   1153
#define OFF_S   1154
#define OFF_OA  901154

typedef __attribute__((ext_vector_type(16))) _Float16 v16h;
typedef __attribute__((ext_vector_type(8)))  _Float16 v8h;
typedef __attribute__((ext_vector_type(4)))  _Float16 v4h;
typedef __attribute__((ext_vector_type(8)))  float    v8f;
typedef __attribute__((ext_vector_type(4)))  float    v4f;

#define LSTRIDE 88   // halves per row: 176 B, 16-B aligned, conflict-free banks

// ---------------------------------------------------------------------------
// Kernel 1: h0 = x @ W1  (N x 64 @ 64 x 128) via WMMA f16->f32
// Block = 256 threads = 8 waves; block covers 128 rows, each wave 16 rows.
// LDS tiles are stored pre-swizzled in WMMA fragment order so each lane's
// fragment is two contiguous 16-B ds_load_b128's.
// ---------------------------------------------------------------------------
__global__ __launch_bounds__(256) void gemm_x_w1(const float* __restrict__ x,
                                                 const float* __restrict__ W1,
                                                 float* __restrict__ h0) {
    __shared__ _Float16 lx[128 * LSTRIDE];   // x tile, fragment-ordered rows
    __shared__ _Float16 lwT[128 * LSTRIDE];  // W1 transposed: [n][k]
    const int t = threadIdx.x;
    const int rowBase = blockIdx.x * 128;

    // --- stage x tile: float4 global loads, packed half4 LDS stores ---
    // row layout (positions): k-groups [0,2,4,6 | 1,3,5,7] of 8
    #pragma unroll
    for (int i = 0; i < 8; ++i) {
        int q = t + i * 256;                  // quad index, 0..2047
        int base = q * 4;
        int r = base >> 6, c = base & 63;     // c multiple of 4
        v4f v = {0.0f, 0.0f, 0.0f, 0.0f};
        int gr = rowBase + r;
        if (gr < Nn) v = *(const v4f*)(x + gr * INc + c);
        int g = c >> 3;
        int pg = (g & 1) * 4 + (g >> 1);
        int p0 = pg * 8 + (c & 7);            // multiple of 4
        v4h hv;
        hv[0] = (_Float16)v[0]; hv[1] = (_Float16)v[1];
        hv[2] = (_Float16)v[2]; hv[3] = (_Float16)v[3];
        *(v4h*)(lx + r * LSTRIDE + p0) = hv;
    }
    // --- stage W1 transposed: lwT[n*LSTRIDE + k] ---
    #pragma unroll
    for (int i = 0; i < 8; ++i) {
        int q = t + i * 256;                  // 0..2047
        int base = q * 4;                     // k*128 + n, n multiple of 4
        int k = base >> 7, n = base & 127;
        v4f v = *(const v4f*)(W1 + base);
        lwT[(n + 0) * LSTRIDE + k] = (_Float16)v[0];
        lwT[(n + 1) * LSTRIDE + k] = (_Float16)v[1];
        lwT[(n + 2) * LSTRIDE + k] = (_Float16)v[2];
        lwT[(n + 3) * LSTRIDE + k] = (_Float16)v[3];
    }
    __syncthreads();

    const int wave = t >> 5;
    const int lane = t & 31;
    const int halfSel = lane >> 4;
    const int l15 = lane & 15;
    const int waveRow = wave * 16;
    const bool waveValid = (rowBase + waveRow) < Nn;   // Nn % 16 == 0 -> uniform

    // A fragments: contiguous in swizzled row
    const int aOff = (waveRow + l15) * LSTRIDE + halfSel * 32;
    v8h a0lo = *(const v8h*)(lx + aOff);
    v8h a0hi = *(const v8h*)(lx + aOff + 8);
    v8h a1lo = *(const v8h*)(lx + aOff + 16);
    v8h a1hi = *(const v8h*)(lx + aOff + 24);
    v16h a0 = __builtin_shufflevector(a0lo, a0hi, 0,1,2,3,4,5,6,7,8,9,10,11,12,13,14,15);
    v16h a1 = __builtin_shufflevector(a1lo, a1hi, 0,1,2,3,4,5,6,7,8,9,10,11,12,13,14,15);

    #pragma unroll
    for (int cT = 0; cT < 8; ++cT) {
        const int n0 = cT * 16;
        // B fragments: lanes 0-15 hold K 0..7,16..23, lanes 16-31 K 8..15,24..31 (per k-step)
        const int bOff = (n0 + l15) * LSTRIDE + halfSel * 8;
        v8h b0lo = *(const v8h*)(lwT + bOff);
        v8h b0hi = *(const v8h*)(lwT + bOff + 16);
        v8h b1lo = *(const v8h*)(lwT + bOff + 32);
        v8h b1hi = *(const v8h*)(lwT + bOff + 48);
        v16h b0  = __builtin_shufflevector(b0lo, b0hi, 0,1,2,3,4,5,6,7,8,9,10,11,12,13,14,15);
        v16h b1v = __builtin_shufflevector(b1lo, b1hi, 0,1,2,3,4,5,6,7,8,9,10,11,12,13,14,15);

        v8f acc = {};
        acc = __builtin_amdgcn_wmma_f32_16x16x32_f16(false, a0, false, b0,
                                                     (short)0, acc, false, false);
        acc = __builtin_amdgcn_wmma_f32_16x16x32_f16(false, a1, false, b1v,
                                                     (short)0, acc, false, false);
        // C/D layout: VGPR v -> M = v + halfSel*8, N = lane&15
        if (waveValid) {
            const int grBase = rowBase + waveRow + halfSel * 8;
            #pragma unroll
            for (int v = 0; v < 8; ++v)
                h0[(grBase + v) * Hc + n0 + l15] = acc[v];
        }
    }
}

// ---------------------------------------------------------------------------
// Kernel 2: deg = 1 (self loop), zero small accumulators [1315 floats]
// ---------------------------------------------------------------------------
__global__ void init_kernel(float* __restrict__ deg, float* __restrict__ accum) {
    int i = blockIdx.x * blockDim.x + threadIdx.x;
    if (i < Nn) deg[i] = 1.0f;
    if (i < 1315) accum[i] = 0.0f;
}

// ---------------------------------------------------------------------------
// Kernel 3: deg[dst] += 1 per edge
// ---------------------------------------------------------------------------
__global__ void deg_kernel(const int* __restrict__ ei, float* __restrict__ deg) {
    int e = blockIdx.x * blockDim.x + threadIdx.x;
    if (e < Ee) atomicAdd(&deg[ei[Ee + e]], 1.0f);
}

// ---------------------------------------------------------------------------
// Kernel 4: agg[n][j] = h0[n][j] / deg[n]   (self-loop term, dinv^2 = 1/deg)
// ---------------------------------------------------------------------------
__global__ void selfloop_kernel(const float* __restrict__ h0,
                                const float* __restrict__ deg,
                                float* __restrict__ agg) {
    int i = blockIdx.x * blockDim.x + threadIdx.x;
    if (i < Nn * Hc) {
        int n = i >> 7;
        agg[i] = h0[i] / deg[n];
    }
}

// ---------------------------------------------------------------------------
// Kernel 5: agg[dst][j] += dinv[src]*dinv[dst]*h0[src][j]. 128 lanes per edge.
// ---------------------------------------------------------------------------
__global__ __launch_bounds__(256) void agg_kernel(const int* __restrict__ ei,
                                                  const float* __restrict__ h0,
                                                  const float* __restrict__ deg,
                                                  float* __restrict__ agg) {
    int t = threadIdx.x;
    int e = blockIdx.x * 2 + (t >> 7);
    int j = t & 127;
    if (e < Ee) {
        int src = ei[e];
        int dst = ei[Ee + e];
        float norm = rsqrtf(deg[src]) * rsqrtf(deg[dst]);
        atomicAdd(&agg[dst * Hc + j], norm * h0[src * Hc + j]);
    }
}

// ---------------------------------------------------------------------------
// Kernel 6: per node (wave-per-node): hx = relu(agg+b1); logits = hx@Wp + bp;
// s = softmax; write s; accumulate x_pool = s^T hx (per-wave LDS, then atomics)
// ---------------------------------------------------------------------------
__global__ __launch_bounds__(256) void softmax_kernel(const float* __restrict__ agg,
                                                      const float* __restrict__ b1,
                                                      const float* __restrict__ Wp,
                                                      const float* __restrict__ bp,
                                                      float* __restrict__ s_out,
                                                      float* __restrict__ xpool_g) {
    __shared__ float lWp[Hc * Kc];
    __shared__ float lb1[Hc];
    __shared__ float lbp[Kc];
    __shared__ float xp[8][Hc * Kc];   // per-wave x_pool accumulators
    const int t = threadIdx.x;
    const int wave = t >> 5, lane = t & 31;

    for (int i = t; i < Hc * Kc; i += 256) lWp[i] = Wp[i];
    if (t < Hc) lb1[t] = b1[t];
    if (t < Kc) lbp[t] = bp[t];
    for (int i = lane; i < Hc * Kc; i += 32) xp[wave][i] = 0.0f;
    __syncthreads();

    const int totalWaves = gridDim.x * 8;
    for (int node = blockIdx.x * 8 + wave; node < Nn; node += totalWaves) {
        float hx[4];
        #pragma unroll
        for (int q = 0; q < 4; ++q) {
            int j = lane + q * 32;
            float v = agg[node * Hc + j] + lb1[j];
            hx[q] = v > 0.0f ? v : 0.0f;
        }
        float p[Kc];
        #pragma unroll
        for (int k = 0; k < Kc; ++k) {
            float a = 0.0f;
            #pragma unroll
            for (int q = 0; q < 4; ++q) a += hx[q] * lWp[(lane + q * 32) * Kc + k];
            p[k] = a;
        }
        #pragma unroll
        for (int k = 0; k < Kc; ++k) {
            #pragma unroll
            for (int off = 16; off >= 1; off >>= 1)
                p[k] += __shfl_xor(p[k], off, 32);
            p[k] += lbp[k];
        }
        float m = p[0];
        #pragma unroll
        for (int k = 1; k < Kc; ++k) m = fmaxf(m, p[k]);
        float sk[Kc];
        float sum = 0.0f;
        #pragma unroll
        for (int k = 0; k < Kc; ++k) { sk[k] = expf(p[k] - m); sum += sk[k]; }
        float inv = 1.0f / sum;
        #pragma unroll
        for (int k = 0; k < Kc; ++k) sk[k] *= inv;

        float val = 0.0f;
        #pragma unroll
        for (int k = 0; k < Kc; ++k) if (lane == k) val = sk[k];
        if (lane < Kc) s_out[node * Kc + lane] = val;

        #pragma unroll
        for (int k = 0; k < Kc; ++k) {
            #pragma unroll
            for (int q = 0; q < 4; ++q)
                xp[wave][k * Hc + lane + q * 32] += sk[k] * hx[q];
        }
    }
    __syncthreads();
    for (int i = t; i < Hc * Kc; i += 256) {
        float s = 0.0f;
        #pragma unroll
        for (int w = 0; w < 8; ++w) s += xp[w][i];
        atomicAdd(&xpool_g[i], s);
    }
}

// ---------------------------------------------------------------------------
// Kernel 7: out_adj[k][l] = sum_e s[src][k]*s[dst][l]
// ---------------------------------------------------------------------------
__global__ __launch_bounds__(256) void outadj_kernel(const int* __restrict__ ei,
                                                     const float* __restrict__ s,
                                                     float* __restrict__ oadj_g) {
    __shared__ float lacc[81];
    const int t = threadIdx.x;
    if (t < 81) lacc[t] = 0.0f;
    __syncthreads();

    float acc[Kc][Kc];
    #pragma unroll
    for (int k = 0; k < Kc; ++k)
        #pragma unroll
        for (int l = 0; l < Kc; ++l) acc[k][l] = 0.0f;

    const int stride = gridDim.x * 256;
    for (int e = blockIdx.x * 256 + t; e < Ee; e += stride) {
        int src = ei[e], dst = ei[Ee + e];
        float a[Kc], b[Kc];
        #pragma unroll
        for (int k = 0; k < Kc; ++k) { a[k] = s[src * Kc + k]; b[k] = s[dst * Kc + k]; }
        #pragma unroll
        for (int k = 0; k < Kc; ++k)
            #pragma unroll
            for (int l = 0; l < Kc; ++l) acc[k][l] += a[k] * b[l];
    }
    #pragma unroll
    for (int k = 0; k < Kc; ++k)
        #pragma unroll
        for (int l = 0; l < Kc; ++l) atomicAdd(&lacc[k * Kc + l], acc[k][l]);
    __syncthreads();
    if (t < 81) atomicAdd(&oadj_g[t], lacc[t]);
}

// ---------------------------------------------------------------------------
// Kernel 8: ss = S^T S (upper triangle) and mincut_den = sum deg_a * |s|^2
// ---------------------------------------------------------------------------
__global__ __launch_bounds__(256) void ssden_kernel(const float* __restrict__ s,
                                                    const float* __restrict__ deg,
                                                    float* __restrict__ ss_g,
                                                    float* __restrict__ den_g) {
    __shared__ float lacc[82];
    const int t = threadIdx.x;
    if (t < 82) lacc[t] = 0.0f;
    __syncthreads();

    float acc[Kc][Kc];
    #pragma unroll
    for (int k = 0; k < Kc; ++k)
        #pragma unroll
        for (int l = k; l < Kc; ++l) acc[k][l] = 0.0f;
    float den = 0.0f;

    const int stride = gridDim.x * 256;
    for (int n = blockIdx.x * 256 + t; n < Nn; n += stride) {
        float a[Kc];
        float s2 = 0.0f;
        #pragma unroll
        for (int k = 0; k < Kc; ++k) { a[k] = s[n * Kc + k]; s2 += a[k] * a[k]; }
        den += (deg[n] - 1.0f) * s2;   // deg_a = in-degree (deg minus self loop)
        #pragma unroll
        for (int k = 0; k < Kc; ++k)
            #pragma unroll
            for (int l = k; l < Kc; ++l) acc[k][l] += a[k] * a[l];
    }
    #pragma unroll
    for (int k = 0; k < Kc; ++k)
        #pragma unroll
        for (int l = k; l < Kc; ++l) atomicAdd(&lacc[k * Kc + l], acc[k][l]);
    atomicAdd(&lacc[81], den);
    __syncthreads();
    if (t < 81) atomicAdd(&ss_g[t], lacc[t]);
    if (t == 81) atomicAdd(den_g, lacc[81]);
}

// ---------------------------------------------------------------------------
// Kernel 9: finalize — mc, o, normalized oa, log_softmax(x_pool). 1 block.
// ---------------------------------------------------------------------------
__global__ __launch_bounds__(256) void finalize_kernel(const float* __restrict__ xpool,
                                                       const float* __restrict__ oadj,
                                                       const float* __restrict__ ssu,
                                                       const float* __restrict__ den,
                                                       float* __restrict__ out) {
    __shared__ float A[81];
    __shared__ float SS[81];
    __shared__ float dis[Kc];
    __shared__ float rowl[Kc];
    const int t = threadIdx.x;

    if (t < 81) {
        A[t] = oadj[t];
        int k = t / Kc, l = t % Kc;
        SS[t] = (l >= k) ? ssu[t] : ssu[l * Kc + k];   // mirror upper triangle
    }
    __syncthreads();

    if (t == 0) {
        float num = 0.0f;
        for (int k = 0; k < Kc; ++k) num += A[k * Kc + k];
        out[OFF_MC] = -(num / (den[0] + EPSf));

        float fn = 0.0f;
        for (int i = 0; i < 81; ++i) fn += SS[i] * SS[i];
        fn = sqrtf(fn);
        float o = 0.0f;
        for (int k = 0; k < Kc; ++k)
            for (int l = 0; l < Kc; ++l) {
                float v = SS[k * Kc + l] / fn - (k == l ? (1.0f / 3.0f) : 0.0f);
                o += v * v;
            }
        out[OFF_O] = sqrtf(o);
    }

    if (t < Kc) {
        float d = 0.0f;
        for (int l = 0; l < Kc; ++l) if (l != t) d += A[t * Kc + l];
        dis[t] = rsqrtf(d + EPSf);
    }
    __syncthreads();
    if (t < 81) {
        int k = t / Kc, l = t % Kc;
        out[OFF_OA + t] = (k == l) ? 0.0f : dis[k] * A[t] * dis[l];
    }

    if (t < Kc) {
        float m = -3.4e38f;
        for (int c = 0; c < Hc; ++c) m = fmaxf(m, xpool[t * Hc + c]);
        float sum = 0.0f;
        for (int c = 0; c < Hc; ++c) sum += expf(xpool[t * Hc + c] - m);
        rowl[t] = m + logf(sum);
    }
    __syncthreads();
    for (int i = t; i < Kc * Hc; i += 256) {
        int k = i >> 7;
        out[OFF_OUT + i] = xpool[i] - rowl[k];
    }
}

// ---------------------------------------------------------------------------
extern "C" void kernel_launch(void* const* d_in, const int* in_sizes, int n_in,
                              void* d_out, int out_size, void* d_ws, size_t ws_size,
                              hipStream_t stream) {
    const float* x  = (const float*)d_in[0];
    const int*   ei = (const int*)d_in[1];
    // d_in[2] = batch (unused, single graph)
    const float* W1 = (const float*)d_in[3];
    const float* b1 = (const float*)d_in[4];
    const float* Wp = (const float*)d_in[5];
    const float* bp = (const float*)d_in[6];
    float* out = (float*)d_out;
    float* ws  = (float*)d_ws;

    float* h0    = ws;                    // N*H = 12,800,000
    float* deg   = ws + 12800000;         // N   = 100,000
    float* agg   = ws + 12900000;         // N*H = 12,800,000
    float* xpool = ws + 25700000;         // 1152
    float* oadj  = xpool + 1152;          // 81
    float* ssu   = oadj + 81;             // 81
    float* den   = ssu + 81;              // 1   (accum region total = 1315)
    float* s     = out + OFF_S;           // [N, K] lives directly in d_out

    gemm_x_w1<<<(Nn + 127) / 128, 256, 0, stream>>>(x, W1, h0);
    init_kernel<<<(Nn + 255) / 256, 256, 0, stream>>>(deg, xpool);
    deg_kernel<<<(Ee + 255) / 256, 256, 0, stream>>>(ei, deg);
    selfloop_kernel<<<(Nn * Hc + 255) / 256, 256, 0, stream>>>(h0, deg, agg);
    agg_kernel<<<(Ee + 1) / 2, 256, 0, stream>>>(ei, h0, deg, agg);
    softmax_kernel<<<1024, 256, 0, stream>>>(agg, b1, Wp, bp, s, xpool);
    outadj_kernel<<<1024, 256, 0, stream>>>(ei, s, oadj);
    ssden_kernel<<<512, 256, 0, stream>>>(s, deg, ssu, den);
    finalize_kernel<<<1, 256, 0, stream>>>(xpool, oadj, ssu, den, out);
}
